// RNNcell_44676249813133
// MI455X (gfx1250) — matrix operand order repaired and checked by reading, not compiled, not run on hardware
//
#include <hip/hip_runtime.h>
#include <hip/hip_fp16.h>
#include <math.h>

// RNN: T=512, B=64, D=H=1024, fp32 reference.
// Strategy: f16 WMMA (16x16x32) with f32 accumulation.
//   phase 0: transpose-convert W_ih, W_hh to f16 [N][K] (K contiguous); convert init h.
//   phase 1: x_proj = input @ W_ih + bias  (big parallel GEMM, f32 out)
//   phase 2: 512 sequential step kernels: h = tanh(x_proj[t] + h @ W_hh)

typedef __attribute__((ext_vector_type(16))) _Float16 v16h;
typedef __attribute__((ext_vector_type(8)))  float    v8f;
typedef __attribute__((ext_vector_type(4)))  float    v4f;
typedef __attribute__((ext_vector_type(4)))  unsigned int v4u;

union V16H { v16h h; v4u u[2]; };

#define RNN_T 512
#define RNN_B 64
#define RNN_D 1024
#define RNN_H 1024

// ---- phase 0: convert+transpose f32 [rows][cols] -> f16 [cols][rows] ----
__global__ void k_cvt_transpose(const float* __restrict__ src,
                                _Float16* __restrict__ dst,
                                int rows, int cols) {
  int idx = blockIdx.x * blockDim.x + threadIdx.x;
  int r = idx / cols;
  int c = idx % cols;
  dst[(size_t)c * rows + r] = (_Float16)src[idx];
}

__global__ void k_cvt(const float* __restrict__ src, _Float16* __restrict__ dst, int n) {
  int i = blockIdx.x * blockDim.x + threadIdx.x;
  if (i < n) dst[i] = (_Float16)src[i];
}

// ---- phase 1: Y[M,N] = X[M,K] @ WT[N,K]^T + bias[N] ----
// One wave per 16x64 output strip: A fragment reused across 4 N-tiles.
__global__ void __launch_bounds__(256) k_xproj(const float* __restrict__ X,
                                               const _Float16* __restrict__ WT,
                                               const float* __restrict__ bias,
                                               float* __restrict__ Y) {
  const int K = RNN_D, N = RNN_H;
  int wid  = blockIdx.x * 8 + (threadIdx.x >> 5);
  int lane = threadIdx.x & 31;
  int mt = wid >> 4;            // 2048 M-tiles
  int nb = (wid & 15) << 6;     // 16 strips of 64 columns
  int m0 = mt << 4;
  int g  = lane >> 4;           // lane half
  int lm = lane & 15;

  const float* arow = X + (size_t)(m0 + lm) * K;
  v8f acc[4] = {v8f{}, v8f{}, v8f{}, v8f{}};

  for (int k0 = 0; k0 < K; k0 += 32) {
    // A fragment: convert f32 -> f16 in registers (4x b128 loads, contiguous)
    v4f a0 = *(const v4f*)(arow + k0 + 8 * g);
    v4f a1 = *(const v4f*)(arow + k0 + 8 * g + 4);
    v4f a2 = *(const v4f*)(arow + k0 + 16 + 8 * g);
    v4f a3 = *(const v4f*)(arow + k0 + 16 + 8 * g + 4);
    v16h a;
#pragma unroll
    for (int j = 0; j < 4; ++j) {
      a[j]      = (_Float16)a0[j];
      a[4 + j]  = (_Float16)a1[j];
      a[8 + j]  = (_Float16)a2[j];
      a[12 + j] = (_Float16)a3[j];
    }
#pragma unroll
    for (int s = 0; s < 4; ++s) {
      const _Float16* bp = WT + (size_t)(nb + s * 16 + lm) * K + k0 + 8 * g;
      V16H b;
      b.u[0] = *(const v4u*)(bp);
      b.u[1] = *(const v4u*)(bp + 16);
      acc[s] = __builtin_amdgcn_wmma_f32_16x16x32_f16(false, a, false, b.h,
                                                      (short)0, acc[s], false, false);
    }
  }

  int mrow = m0 + 8 * g;
#pragma unroll
  for (int s = 0; s < 4; ++s) {
    int n = nb + s * 16 + lm;
    float bv = bias[n];
#pragma unroll
    for (int r = 0; r < 8; ++r)
      Y[(size_t)(mrow + r) * N + n] = acc[s][r] + bv;
  }
}

// ---- phase 2: one timestep: hout = tanh(xp + hin @ WhhT^T) ----
// One wave per 16x16 tile; 256 waves; chained WMMA over K=1024.
__global__ void __launch_bounds__(256) k_step(const _Float16* __restrict__ hin,
                                              const _Float16* __restrict__ WT,
                                              const float* __restrict__ xp,
                                              _Float16* __restrict__ hout,
                                              float* __restrict__ out0,
                                              float* __restrict__ out1) {
  const int K = RNN_H, N = RNN_H;
  int wid  = blockIdx.x * 8 + (threadIdx.x >> 5);  // 0..255
  int lane = threadIdx.x & 31;
  int mt = wid & 3, nt = wid >> 2;
  int m0 = mt << 4, n0 = nt << 4;
  int g  = lane >> 4;
  int lm = lane & 15;

  const _Float16* arow = hin + (size_t)(m0 + lm) * K;
  const _Float16* bcol = WT  + (size_t)(n0 + lm) * K;
  v8f acc = {};

  for (int k0 = 0; k0 < K; k0 += 32) {
    V16H a, b;
    a.u[0] = *(const v4u*)(arow + k0 + 8 * g);
    a.u[1] = *(const v4u*)(arow + k0 + 16 + 8 * g);
    b.u[0] = *(const v4u*)(bcol + k0 + 8 * g);
    b.u[1] = *(const v4u*)(bcol + k0 + 16 + 8 * g);
    acc = __builtin_amdgcn_wmma_f32_16x16x32_f16(false, a.h, false, b.h,
                                                 (short)0, acc, false, false);
  }

  int n = n0 + lm;
  int mrow = m0 + 8 * g;
#pragma unroll
  for (int r = 0; r < 8; ++r) {
    size_t idx = (size_t)(mrow + r) * N + n;
    float v = tanhf(acc[r] + xp[idx]);
    hout[idx] = (_Float16)v;
    if (out0) {  // last timestep: emit f32 final state to both tuple slots
      out0[idx] = v;
      out1[idx] = v;
    }
  }
}

extern "C" void kernel_launch(void* const* d_in, const int* in_sizes, int n_in,
                              void* d_out, int out_size, void* d_ws, size_t ws_size,
                              hipStream_t stream) {
  const int T = RNN_T, B = RNN_B, D = RNN_D, H = RNN_H;
  const float* input      = (const float*)d_in[0];  // [T,B,D]
  const float* init_state = (const float*)d_in[1];  // [B,H]
  const float* W_ih       = (const float*)d_in[2];  // [D,H]
  const float* W_hh       = (const float*)d_in[3];  // [H,H]
  const float* bias_hh    = (const float*)d_in[4];  // [H]

  char* ws = (char*)d_ws;
  size_t off = 0;
  float* xproj = (float*)(ws + off);      off += (size_t)T * B * H * sizeof(float);
  _Float16* WihT = (_Float16*)(ws + off); off += (size_t)D * H * sizeof(_Float16);
  _Float16* WhhT = (_Float16*)(ws + off); off += (size_t)H * H * sizeof(_Float16);
  _Float16* hA   = (_Float16*)(ws + off); off += (size_t)B * H * sizeof(_Float16);
  _Float16* hB   = (_Float16*)(ws + off); off += (size_t)B * H * sizeof(_Float16);
  (void)ws_size; (void)in_sizes; (void)n_in; (void)out_size;

  // phase 0: weight transpose-convert + h0 convert
  k_cvt_transpose<<<(D * H) / 256, 256, 0, stream>>>(W_ih, WihT, D, H);
  k_cvt_transpose<<<(H * H) / 256, 256, 0, stream>>>(W_hh, WhhT, H, H);
  k_cvt<<<(B * H) / 256, 256, 0, stream>>>(init_state, hA, B * H);

  // phase 1: x_proj = input @ W_ih + bias   (M=T*B=32768 rows)
  // waves = (M/16) * (N/64) = 2048*16 = 32768 -> 4096 blocks of 8 waves
  k_xproj<<<4096, 256, 0, stream>>>(input, WihT, bias_hh, xproj);

  // phase 2: sequential scan, 512 launches (graph-captured by harness)
  float* out = (float*)d_out;
  for (int t = 0; t < T; ++t) {
    _Float16* hin  = (t & 1) ? hB : hA;
    _Float16* hout = (t & 1) ? hA : hB;
    bool last = (t == T - 1);
    k_step<<<32, 256, 0, stream>>>(hin, WhhT, xproj + (size_t)t * B * H, hout,
                                   last ? out : nullptr,
                                   last ? out + (size_t)B * H : nullptr);
  }
}
